// portHNN_dgl_43379169689825
// MI455X (gfx1250) — compile-verified, matrix operand order checked by reading, not run on hardware
//
#include <hip/hip_runtime.h>

// portHNN on MI455X (gfx1250, wave32, WMMA).
// Strategy: all [N,128] node tables are L2-resident (<=25.6MB vs 192MB L2).
// Edge-side work (the 79 GFLOP 3-layer edge MLP) is fused into one kernel:
// gather p1[src]+p2[dst] -> 3x WMMA GEMM layers in LDS -> *K_node[src] ->
// f32 atomic scatter. No [E,128] tensor ever touches HBM.
// v2: weights staged in LDS (ds_load B-frags instead of global loads),
//     zero-init accumulators (inline-0 C operand), bias in epilogue,
//     global_prefetch of next-layer weights.

#define LDH 136          // LDS row pitch in halves (272B = 16B aligned, bank-skewed)
#define TILE_ELEMS (128 * LDH)
#define SMEM_BYTES (2 * TILE_ELEMS * 2)  // tile + weight panel, f16

typedef __attribute__((ext_vector_type(16))) _Float16 v16h;
typedef __attribute__((ext_vector_type(8)))  _Float16 v8h;
typedef __attribute__((ext_vector_type(8)))  float    v8f;

// ---------------------------------------------------------------------------
// WMMA fragment load, 16-bit A (16x32) / mirrored B layout (ISA 7.12.2):
// lane l%16 selects the row; lane group g = lane/16 selects K sub-bands.
// halves 0..7  = [kb + 8g .. kb + 8g + 7]
// halves 8..15 = [kb + 16 + 8g .. +7]
// Both 16B-aligned b128 loads (row pitch 136 halves = 272B).
// ---------------------------------------------------------------------------
__device__ __forceinline__ v16h frag16(const _Float16* rowp, int kb, int g) {
  const _Float16* p = rowp + kb + (g << 3);
  v8h lo = *(const v8h*)(p);
  v8h hi = *(const v8h*)(p + 16);
  v16h r;
#pragma unroll
  for (int i = 0; i < 8; ++i) { r[i] = lo[i]; r[8 + i] = hi[i]; }
  return r;
}

// cooperative copy of a 128x128 f16 weight matrix (global, L2-hot) into LDS
__device__ __forceinline__ void copy_w_to_lds(const _Float16* __restrict__ Wg,
                                              _Float16* Wl) {
  for (int idx = threadIdx.x; idx < 128 * 16; idx += 256) {
    const int row = idx >> 4, c8 = (idx & 15) << 3;
    *(v8h*)(Wl + row * LDH + c8) = *(const v8h*)(Wg + (size_t)row * 128 + c8);
  }
  __syncthreads();
}

// ---------------------------------------------------------------------------
// One 128x128 layer: out[m][n] = act( sum_k tile[m][k] * Wl[n][k] + bias[n] )
// tile, Wl: LDS f16 [128][LDH]. tile updated in place when writeLds.
// 8 waves; wave w owns rows [16w,16w+16); 8 acc tiles live in VGPRs so the
// in-place LDS update is hazard-free (sync between read and write phases).
// Accumulators start at inline 0; bias folded into the epilogue.
// ACT: 0=none 1=tanh 2=relu
// ---------------------------------------------------------------------------
template <int ACT>
__device__ __forceinline__ void gemm128_layer(
    _Float16* tile, const _Float16* Wl, const float* __restrict__ biasG,
    float* outG32, _Float16* outG16, int rowsValid, bool writeLds) {
  const int lane = threadIdx.x & 31;
  const int wave = threadIdx.x >> 5;
  const int g = lane >> 4, l = lane & 15;
  const _Float16* arow = tile + (wave * 16 + l) * LDH;

  v8f acc[8];
#pragma unroll
  for (int tn = 0; tn < 8; ++tn) {
    v8f c = {0.f, 0.f, 0.f, 0.f, 0.f, 0.f, 0.f, 0.f};
    const _Float16* brow = Wl + (tn * 16 + l) * LDH;
#pragma unroll
    for (int kb = 0; kb < 128; kb += 32) {
      v16h a = frag16(arow, kb, g);
      v16h b = frag16(brow, kb, g);
      c = __builtin_amdgcn_wmma_f32_16x16x32_f16(
          false, a, false, b, (short)0, c, false, false);
    }
    acc[tn] = c;
  }
  __syncthreads();  // all reads of tile done before in-place overwrite
#pragma unroll
  for (int tn = 0; tn < 8; ++tn) {
    const int n = tn * 16 + l;
    const float bn = biasG[n];
#pragma unroll
    for (int r = 0; r < 8; ++r) {
      const int m = wave * 16 + r + 8 * g;
      float v = acc[tn][r] + bn;
      if (ACT == 1) v = tanhf(v);
      if (ACT == 2) v = fmaxf(v, 0.0f);
      if (writeLds) tile[m * LDH + n] = (_Float16)v;
      if (m < rowsValid) {
        if (outG32) outG32[(size_t)m * 128 + n] = v;
        if (outG16) outG16[(size_t)m * 128 + n] = (_Float16)v;
      }
    }
  }
  __syncthreads();
}

// load a [128,128] f32 strip into LDS tile as f16 (zero-pad past N)
__device__ __forceinline__ void load_tile_f32(const float* __restrict__ in,
                                              _Float16* tile, int base, int N) {
  for (int idx = threadIdx.x; idx < 128 * 16; idx += 256) {
    const int row = idx >> 4, c8 = (idx & 15) << 3;
    const int gr = base + row;
    v8h h;
    if (gr < N) {
      const float* p = in + (size_t)gr * 128 + c8;
#pragma unroll
      for (int j = 0; j < 8; ++j) h[j] = (_Float16)p[j];
    } else {
#pragma unroll
      for (int j = 0; j < 8; ++j) h[j] = (_Float16)0.0f;
    }
    *(v8h*)(tile + row * LDH + c8) = h;
  }
  __syncthreads();
}

// prefetch one 32KB weight matrix toward the WGP (global_prefetch_b8)
__device__ __forceinline__ void prefetch_w(const _Float16* Wg) {
  if (threadIdx.x < 128)  // 128 x 256B cachelines
    __builtin_prefetch(Wg + (size_t)threadIdx.x * 128, 0, 1);
}

// ------------------------------ small utils --------------------------------
__global__ void k_f32_to_f16(const float* __restrict__ in,
                             _Float16* __restrict__ out, int n) {
  int t = blockIdx.x * blockDim.x + threadIdx.x;
  if (t < n) out[t] = (_Float16)in[t];
}

__global__ void k_zero(float* __restrict__ p, int n) {
  int t = blockIdx.x * blockDim.x + threadIdx.x;
  if (t < n) p[t] = 0.0f;
}

// ------------------------- stage 1: node encoders --------------------------
__global__ void k_encoders(const float* __restrict__ x,
                           const _Float16* Wk, const float* bk,
                           const _Float16* W1, const float* b1,
                           const _Float16* W2, const float* b2,
                           float* __restrict__ h1, _Float16* __restrict__ p1,
                           _Float16* __restrict__ p2, int N) {
  extern __shared__ __align__(16) _Float16 smem[];
  _Float16* tile = smem;
  _Float16* Wl = smem + TILE_ELEMS;
  const int base = blockIdx.x * 128;
  const int rv = min(128, N - base);
  prefetch_w(W1);
  prefetch_w(W2);
  load_tile_f32(x, tile, base, N);
  copy_w_to_lds(Wk, Wl);
  gemm128_layer<0>(tile, Wl, bk, h1 + (size_t)base * 128, nullptr, rv, false);
  copy_w_to_lds(W1, Wl);
  gemm128_layer<0>(tile, Wl, b1, nullptr, p1 + (size_t)base * 128, rv, false);
  copy_w_to_lds(W2, Wl);
  gemm128_layer<0>(tile, Wl, b2, nullptr, p2 + (size_t)base * 128, rv, false);
}

// --------------------- edge scatter: acc[dst] += rows[src] -----------------
__global__ void k_scatter_rows(const float* __restrict__ rows,
                               const int* __restrict__ src,
                               const int* __restrict__ dst,
                               float* __restrict__ acc, int E) {
  const int t = blockIdx.x * blockDim.x + threadIdx.x;
  const int e = t >> 5;
  if (e >= E) return;
  const int c = (t & 31) << 2;
  const int s = src[e], d = dst[e];
  const float4 v = *(const float4*)(rows + (size_t)s * 128 + c);
  float* a = acc + (size_t)d * 128 + c;
  atomicAdd(a + 0, v.x);
  atomicAdd(a + 1, v.y);
  atomicAdd(a + 2, v.z);
  atomicAdd(a + 3, v.w);
}

// ------------------------ stage 2: K MLP on nodes --------------------------
__global__ void k_kmlp(const float* __restrict__ h1agg,
                       const _Float16* W1, const float* b1,
                       const _Float16* W2, const float* b2,
                       const _Float16* W3, const float* b3,
                       _Float16* __restrict__ kh, int N) {
  extern __shared__ __align__(16) _Float16 smem[];
  _Float16* tile = smem;
  _Float16* Wl = smem + TILE_ELEMS;
  const int base = blockIdx.x * 128;
  const int rv = min(128, N - base);
  prefetch_w(W2);
  prefetch_w(W3);
  load_tile_f32(h1agg, tile, base, N);
  copy_w_to_lds(W1, Wl);
  gemm128_layer<1>(tile, Wl, b1, nullptr, nullptr, rv, true);   // tanh
  copy_w_to_lds(W2, Wl);
  gemm128_layer<2>(tile, Wl, b2, nullptr, nullptr, rv, true);   // relu
  copy_w_to_lds(W3, Wl);
  gemm128_layer<0>(tile, Wl, b3, nullptr, kh + (size_t)base * 128, rv, false);
}

// --------------- stage 3: fused edge megakernel (U MLP + scatter) ----------
__global__ void k_edge(const int* __restrict__ src, const int* __restrict__ dst,
                       const _Float16* __restrict__ p1h,
                       const _Float16* __restrict__ p2h,
                       const _Float16* __restrict__ kh,
                       const _Float16* W1, const float* b1,
                       const _Float16* W2, const float* b2,
                       const _Float16* W3, const float* b3,
                       float* __restrict__ Enew, int E) {
  extern __shared__ __align__(16) _Float16 smem[];
  _Float16* tile = smem;
  _Float16* Wl = smem + TILE_ELEMS;
  const int ebase = blockIdx.x * 128;
  prefetch_w(W2);
  prefetch_w(W3);
  // build e_in = p1[src] + p2[dst] (both L2-resident f16 tables)
  for (int idx = threadIdx.x; idx < 128 * 16; idx += 256) {
    const int row = idx >> 4, c8 = (idx & 15) << 3;
    const int e = ebase + row;
    v8h h;
    if (e < E) {
      const int s = src[e], d = dst[e];
      v8h a = *(const v8h*)(p1h + (size_t)s * 128 + c8);
      v8h b = *(const v8h*)(p2h + (size_t)d * 128 + c8);
      h = a + b;
    } else {
#pragma unroll
      for (int j = 0; j < 8; ++j) h[j] = (_Float16)0.0f;
    }
    *(v8h*)(tile + row * LDH + c8) = h;
  }
  __syncthreads();
  copy_w_to_lds(W1, Wl);
  gemm128_layer<1>(tile, Wl, b1, nullptr, nullptr, 128, true);  // tanh
  copy_w_to_lds(W2, Wl);
  gemm128_layer<2>(tile, Wl, b2, nullptr, nullptr, 128, true);  // relu
  copy_w_to_lds(W3, Wl);
  gemm128_layer<0>(tile, Wl, b3, nullptr, nullptr, 128, true);  // U_edge in LDS
  // E_new[dst] += K_node[src] * U_edge   (f32 atomics into L2-resident table)
  for (int idx = threadIdx.x; idx < 128 * 16; idx += 256) {
    const int row = idx >> 4, c8 = (idx & 15) << 3;
    const int e = ebase + row;
    if (e >= E) continue;
    const int s = src[e], d = dst[e];
    v8h u = *(const v8h*)(tile + row * LDH + c8);
    v8h k = *(const v8h*)(kh + (size_t)s * 128 + c8);
    float* a = Enew + (size_t)d * 128 + c8;
#pragma unroll
    for (int j = 0; j < 8; ++j) atomicAdd(a + j, (float)u[j] * (float)k[j]);
  }
}

// ----------------- stage 4: dH = Enew@H^T+b ; dD = dH@D^T+b ----------------
__global__ void k_hd(const float* __restrict__ Enew,
                     const _Float16* Wh, const float* bh,
                     const _Float16* Wd, const float* bd,
                     float* __restrict__ dH, float* __restrict__ dD, int N) {
  extern __shared__ __align__(16) _Float16 smem[];
  _Float16* tile = smem;
  _Float16* Wl = smem + TILE_ELEMS;
  const int base = blockIdx.x * 128;
  const int rv = min(128, N - base);
  prefetch_w(Wd);
  load_tile_f32(Enew, tile, base, N);
  copy_w_to_lds(Wh, Wl);
  gemm128_layer<0>(tile, Wl, bh, dH + (size_t)base * 128, nullptr, rv, true);
  copy_w_to_lds(Wd, Wl);
  gemm128_layer<0>(tile, Wl, bd, dD + (size_t)base * 128, nullptr, rv, false);
}

// ------------- stage 5: out = [dH[:,64:], -dH[:,:64]] - d_acc --------------
__global__ void k_final(const float* __restrict__ dH,
                        const float* __restrict__ dacc,
                        float* __restrict__ out, int total) {
  const int t = blockIdx.x * blockDim.x + threadIdx.x;
  if (t >= total) return;
  const int j = t & 127;
  const int row = t >> 7;
  const float v = (j < 64) ? dH[(size_t)row * 128 + 64 + j]
                           : -dH[(size_t)row * 128 + (j - 64)];
  out[t] = v - dacc[t];
}

// ---------------------------------------------------------------------------
extern "C" void kernel_launch(void* const* d_in, const int* in_sizes, int n_in,
                              void* d_out, int out_size, void* d_ws,
                              size_t ws_size, hipStream_t stream) {
  const float* x = (const float*)d_in[0];
  const int* src = (const int*)d_in[1];
  const int* dst = (const int*)d_in[2];
  const float* Wf[11] = {
      (const float*)d_in[3],  (const float*)d_in[5],  (const float*)d_in[7],
      (const float*)d_in[9],  (const float*)d_in[11], (const float*)d_in[13],
      (const float*)d_in[15], (const float*)d_in[17], (const float*)d_in[19],
      (const float*)d_in[21], (const float*)d_in[23]};
  const float* encK_b = (const float*)d_in[4];
  const float* encP1_b = (const float*)d_in[6];
  const float* encP2_b = (const float*)d_in[8];
  const float* K_b1 = (const float*)d_in[10];
  const float* K_b2 = (const float*)d_in[12];
  const float* K_b3 = (const float*)d_in[14];
  const float* U_b1 = (const float*)d_in[16];
  const float* U_b2 = (const float*)d_in[18];
  const float* U_b3 = (const float*)d_in[20];
  const float* H_b = (const float*)d_in[22];
  const float* D_b = (const float*)d_in[24];

  const int N = in_sizes[0] / 128;
  const int E = in_sizes[1];
  const int NF = N * 128;

  // ---- workspace carve (256B aligned) ----
  char* ws = (char*)d_ws;
  size_t off = 0;
  auto take = [&](size_t bytes) -> char* {
    char* p = ws + off;
    off = (off + bytes + 255) & ~(size_t)255;
    return p;
  };
  _Float16* Wh16[11];
  for (int i = 0; i < 11; ++i) Wh16[i] = (_Float16*)take(16384 * 2);
  _Float16* p1h = (_Float16*)take((size_t)NF * 2);
  _Float16* p2h = (_Float16*)take((size_t)NF * 2);
  _Float16* kh = (_Float16*)take((size_t)NF * 2);
  float* h1 = (float*)take((size_t)NF * 4);     // later reused as dH
  float* h1agg = (float*)take((size_t)NF * 4);  // later reused as dD
  float* Enew = (float*)take((size_t)NF * 4);
  float* dacc = (float*)take((size_t)NF * 4);
  float* dH = h1;     // h1 dead after scatter stage
  float* dD = h1agg;  // h1_agg dead after K-MLP

  // ---- weight f32->f16 conversions (tiny, L2-hot afterwards) ----
  for (int i = 0; i < 11; ++i)
    k_f32_to_f16<<<64, 256, 0, stream>>>(Wf[i], Wh16[i], 16384);

  // ---- zero accumulators (needed every call: harness replays the graph) ----
  const int zb = (NF + 255) / 256;
  k_zero<<<zb, 256, 0, stream>>>(h1agg, NF);
  k_zero<<<zb, 256, 0, stream>>>(Enew, NF);
  k_zero<<<zb, 256, 0, stream>>>(dacc, NF);

  const int nb = (N + 127) / 128;
  const int eb = (E + 127) / 128;
  const int sb = (int)(((size_t)E * 32 + 255) / 256);

  // stage 1: node encoders (h1 f32; p1,p2 f16)
  k_encoders<<<nb, 256, SMEM_BYTES, stream>>>(x, Wh16[0], encK_b, Wh16[1],
                                              encP1_b, Wh16[2], encP2_b, h1,
                                              p1h, p2h, N);
  // stage 1b: h1_agg[dst] += h1[src]
  k_scatter_rows<<<sb, 256, 0, stream>>>(h1, src, dst, h1agg, E);
  // stage 2: K_node = MLP(h1_agg)
  k_kmlp<<<nb, 256, SMEM_BYTES, stream>>>(h1agg, Wh16[3], K_b1, Wh16[4], K_b2,
                                          Wh16[5], K_b3, kh, N);
  // stage 3: fused edge MLP + K*U scatter into E_new
  k_edge<<<eb, 256, SMEM_BYTES, stream>>>(src, dst, p1h, p2h, kh, Wh16[6],
                                          U_b1, Wh16[7], U_b2, Wh16[8], U_b3,
                                          Enew, E);
  // stage 4: dH and dD
  k_hd<<<nb, 256, SMEM_BYTES, stream>>>(Enew, Wh16[9], H_b, Wh16[10], D_b, dH,
                                        dD, N);
  // stage 4b: d_acc[dst] += dD[src]
  k_scatter_rows<<<sb, 256, 0, stream>>>(dD, src, dst, dacc, E);
  // stage 5: symplectic combine
  k_final<<<zb, 256, 0, stream>>>(dH, dacc, (float*)d_out, NF);
}
